// RNN_decoder_15152644621132
// MI455X (gfx1250) — matrix operand order repaired and checked by reading, not compiled, not run on hardware
//
#include <hip/hip_runtime.h>
#include <hip/hip_bf16.h>
#include <math.h>

// ---------------- problem constants ----------------
constexpr int kB     = 16;
constexpr int kT     = 128;
constexpr int kS     = 128;
constexpr int kHID   = 512;
constexpr int kEMB   = 512;
constexpr int kCTX   = 1024;          // 2*HID
constexpr int kVOCAB = 32000;
constexpr int kIN0   = kEMB + kCTX;   // 1536, GRU0 input width
constexpr int kGATES = 3 * kHID;      // 1536

// ---------------- vector types ----------------
typedef __attribute__((ext_vector_type(16))) __bf16 v16bf_t;
typedef __attribute__((ext_vector_type(8)))  __bf16 v8bf_t;
typedef __attribute__((ext_vector_type(8)))  float  v8f_t;
typedef __attribute__((ext_vector_type(4)))  float  v4f_t;

// ---------------- WMMA fragment loaders ----------------
// A (16x32 bf16, MxK): lane holds row m=l&15; element j <-> K = 16*half + j
__device__ inline v16bf_t load_afrag(const __bf16* __restrict__ row, int kb, int half) {
  v8bf_t lo = *(const v8bf_t*)(row + kb + 16 * half);
  v8bf_t hi = *(const v8bf_t*)(row + kb + 16 * half + 8);
  v16bf_t f;
#pragma unroll
  for (int i = 0; i < 8; ++i) { f[i] = lo[i]; f[i + 8] = hi[i]; }
  return f;
}
__device__ inline v16bf_t load_afrag(const float* __restrict__ row, int kb, int half) {
  const float* p0 = row + kb + 16 * half;
  const float* p1 = p0 + 8;
  v4f_t a0 = *(const v4f_t*)p0, a1 = *(const v4f_t*)(p0 + 4);
  v4f_t b0 = *(const v4f_t*)p1, b1 = *(const v4f_t*)(p1 + 4);
  v16bf_t f;
#pragma unroll
  for (int i = 0; i < 4; ++i) {
    f[i]      = (__bf16)a0[i];
    f[i + 4]  = (__bf16)a1[i];
    f[i + 8]  = (__bf16)b0[i];
    f[i + 12] = (__bf16)b1[i];
  }
  return f;
}
// B (32x16 bf16, KxN) from row-major W[N,K]: lane holds col n=l&15; element j <-> K = 16*half + j
__device__ inline v16bf_t load_bfrag(const __bf16* __restrict__ row, int kb, int half) {
  return *(const v16bf_t*)(row + kb + 16 * half);
}
__device__ inline v16bf_t load_bfrag(const float* __restrict__ row, int kb, int half) {
  const float* p = row + kb + 16 * half;
  v16bf_t f;
#pragma unroll
  for (int i = 0; i < 16; i += 4) {
    v4f_t v = *(const v4f_t*)(p + i);
    f[i] = (__bf16)v[0]; f[i + 1] = (__bf16)v[1];
    f[i + 2] = (__bf16)v[2]; f[i + 3] = (__bf16)v[3];
  }
  return f;
}

#define WMMA_BF16(a, b, c) \
  __builtin_amdgcn_wmma_f32_16x16x32_bf16(false, (a), false, (b), (short)0, (c), false, false)

__device__ inline float sigmoidf_(float v) { return 1.0f / (1.0f + __expf(-v)); }

// ---------------- kernels ----------------

// f32 -> bf16 weight conversion (grid-stride)
__global__ void cvt_bf16_kernel(const float* __restrict__ src, __bf16* __restrict__ dst, size_t n) {
  size_t i = (size_t)blockIdx.x * blockDim.x + threadIdx.x;
  size_t st = (size_t)gridDim.x * blockDim.x;
  for (; i < n; i += st) dst[i] = (__bf16)src[i];
}

// zero the recurrent state
__global__ void init_state_kernel(float* h0f, float* h1f, __bf16* h0b, __bf16* h1b) {
  int i = blockIdx.x * blockDim.x + threadIdx.x;
  if (i < kB * kHID) {
    h0f[i] = 0.0f; h1f[i] = 0.0f;
    h0b[i] = (__bf16)0.0f; h1b[i] = (__bf16)0.0f;
  }
}

// q = tanh([emb[tok] ; h0] @ attn_w^T + attn_b)   -> qf [16, CTX] f32
template <typename TB>
__global__ void qgemm_kernel(const int* __restrict__ x, int t,
                             const float* __restrict__ emb,
                             const TB* __restrict__ attnw,
                             const float* __restrict__ attn_b,
                             const __bf16* __restrict__ h0b,
                             float* __restrict__ qf) {
  int lane = threadIdx.x & 31, wid = threadIdx.x >> 5;
  int tile = blockIdx.x * (blockDim.x >> 5) + wid;
  int n0 = tile * 16;
  int mn = lane & 15, half = lane >> 4;
  int tok = x[mn * kT + t];
  const float*  erow = emb + (size_t)tok * kEMB;
  const __bf16* hrow = h0b + mn * kHID;
  const TB*     wrow = attnw + (size_t)(n0 + mn) * (kEMB + kHID);
  v8f_t acc = {};
  for (int kb = 0; kb < kEMB; kb += 32)
    acc = WMMA_BF16(load_afrag(erow, kb, half), load_bfrag(wrow, kb, half), acc);
  for (int kb = 0; kb < kHID; kb += 32)
    acc = WMMA_BF16(load_afrag(hrow, kb, half), load_bfrag(wrow, kEMB + kb, half), acc);
  int n = n0 + mn;
  float bn = attn_b[n];
#pragma unroll
  for (int r = 0; r < 8; ++r) {
    int mm = r + 8 * half;
    qf[mm * kCTX + n] = tanhf(acc[r] + bn);
  }
}

// per-batch attention: scores -> softmax -> av; assemble inp = [emb[tok] ; av] in bf16
__global__ void attn_kernel(const int* __restrict__ x, int t,
                            const float* __restrict__ context,
                            const float* __restrict__ qf,
                            const float* __restrict__ emb,
                            __bf16* __restrict__ inpb) {
  __shared__ float sc[kS];
  __shared__ float smax, ssum;
  int b = blockIdx.x;
  int tid = threadIdx.x;
  int lane = tid & 31, wid = tid >> 5;
  const float* qrow = qf + b * kCTX;
  for (int s = wid; s < kS; s += 8) {
    const float* crow = context + ((size_t)b * kS + s) * kCTX;
    float p = 0.0f;
    for (int e = lane; e < kCTX; e += 32) p += crow[e] * qrow[e];
#pragma unroll
    for (int m = 16; m >= 1; m >>= 1) p += __shfl_xor(p, m, 32);
    if (lane == 0) sc[s] = p;
  }
  __syncthreads();
  if (tid == 0) {
    float mx = sc[0];
    for (int i = 1; i < kS; ++i) mx = fmaxf(mx, sc[i]);
    float sm = 0.0f;
    for (int i = 0; i < kS; ++i) sm += __expf(sc[i] - mx);
    smax = mx; ssum = sm;
  }
  __syncthreads();
  if (tid < kS) sc[tid] = __expf(sc[tid] - smax) / ssum;
  __syncthreads();
  for (int e = tid; e < kCTX; e += blockDim.x) {
    float av = 0.0f;
    const float* cp = context + (size_t)b * kS * kCTX + e;
    for (int s = 0; s < kS; ++s) av += sc[s] * cp[(size_t)s * kCTX];
    inpb[b * kIN0 + kEMB + e] = (__bf16)av;
  }
  int tok = x[b * kT + t];
  const float* erow = emb + (size_t)tok * kEMB;
  for (int k = tid; k < kEMB; k += blockDim.x) inpb[b * kIN0 + k] = (__bf16)erow[k];
}

// gi = Xi @ Wih^T + b_ih ; gh = H @ Whh^T + b_hh   (both [16, 1536] f32)
template <typename TB>
__global__ void gru_gates_kernel(const __bf16* __restrict__ Xi, int Ki,
                                 const TB* __restrict__ Wih, const float* __restrict__ bih,
                                 const __bf16* __restrict__ H,
                                 const TB* __restrict__ Whh, const float* __restrict__ bhh,
                                 float* __restrict__ gi, float* __restrict__ gh) {
  int lane = threadIdx.x & 31, wid = threadIdx.x >> 5;
  int tile = blockIdx.x * (blockDim.x >> 5) + wid;
  int n0 = tile * 16;
  int mn = lane & 15, half = lane >> 4;
  const __bf16* xrow = Xi + (size_t)mn * Ki;
  const TB*     wir  = Wih + (size_t)(n0 + mn) * Ki;
  v8f_t ai = {};
  for (int kb = 0; kb < Ki; kb += 32)
    ai = WMMA_BF16(load_afrag(xrow, kb, half), load_bfrag(wir, kb, half), ai);
  const __bf16* hrow = H + (size_t)mn * kHID;
  const TB*     whr  = Whh + (size_t)(n0 + mn) * kHID;
  v8f_t ah = {};
  for (int kb = 0; kb < kHID; kb += 32)
    ah = WMMA_BF16(load_afrag(hrow, kb, half), load_bfrag(whr, kb, half), ah);
  int n = n0 + mn;
  float bi = bih[n], bh = bhh[n];
#pragma unroll
  for (int r = 0; r < 8; ++r) {
    int mm = r + 8 * half;
    gi[mm * kGATES + n] = ai[r] + bi;
    gh[mm * kGATES + n] = ah[r] + bh;
  }
}

// GRU gate nonlinearity + state update (f32 state, bf16 shadow); optional masked top output
__global__ void gru_combine_kernel(const float* __restrict__ gi, const float* __restrict__ gh,
                                   float* __restrict__ hf, __bf16* __restrict__ hb,
                                   __bf16* __restrict__ outb,
                                   const int* __restrict__ x, int t) {
  int idx = blockIdx.x * blockDim.x + threadIdx.x;
  if (idx >= kB * kHID) return;
  int m = idx >> 9;      // / kHID
  int j = idx & (kHID - 1);
  const float* gim = gi + m * kGATES;
  const float* ghm = gh + m * kGATES;
  float r = sigmoidf_(gim[j] + ghm[j]);
  float z = sigmoidf_(gim[kHID + j] + ghm[kHID + j]);
  float n = tanhf(gim[2 * kHID + j] + r * ghm[2 * kHID + j]);
  float h = (1.0f - z) * n + z * hf[idx];
  hf[idx] = h;
  hb[idx] = (__bf16)h;
  if (outb != nullptr) {
    float mask = (x[m * kT + t] != 0) ? 1.0f : 0.0f;
    outb[idx] = (__bf16)(h * mask);
  }
}

// logits[:, t, :] = outb @ w_out^T + b_out   ([16,512] x [512,32000])
template <typename TB>
__global__ void logits_kernel(const __bf16* __restrict__ outb,
                              const TB* __restrict__ wout,
                              const float* __restrict__ bout,
                              float* __restrict__ dout, int t) {
  int lane = threadIdx.x & 31, wid = threadIdx.x >> 5;
  int tile = blockIdx.x * (blockDim.x >> 5) + wid;
  int n0 = tile * 16;
  int mn = lane & 15, half = lane >> 4;
  const __bf16* arow = outb + mn * kHID;
  const TB*     brow = wout + (size_t)(n0 + mn) * kHID;
  v8f_t acc = {};
  for (int kb = 0; kb < kHID; kb += 32)
    acc = WMMA_BF16(load_afrag(arow, kb, half), load_bfrag(brow, kb, half), acc);
  int n = n0 + mn;
  float bn = bout[n];
#pragma unroll
  for (int r = 0; r < 8; ++r) {
    int mm = r + 8 * half;
    dout[(size_t)mm * kT * kVOCAB + (size_t)t * kVOCAB + n] = acc[r] + bn;
  }
}

// ---------------- host side ----------------
struct Scratch {
  float *h0f, *h1f, *qf, *gi0, *gh0, *gi1, *gh1;
  __bf16 *h0b, *h1b, *inpb, *outb;
};

template <typename TB>
static void run_model(const int* x, const float* context, const float* emb,
                      const TB* attnw, const float* attn_b,
                      const TB* wih0, const float* bih0, const TB* whh0, const float* bhh0,
                      const TB* wih1, const float* bih1, const TB* whh1, const float* bhh1,
                      const TB* wout, const float* bout,
                      const Scratch& s, float* dout, hipStream_t stream) {
  init_state_kernel<<<32, 256, 0, stream>>>(s.h0f, s.h1f, s.h0b, s.h1b);
  for (int t = 0; t < kT; ++t) {
    qgemm_kernel<TB><<<kCTX / 16 / 8, 256, 0, stream>>>(x, t, emb, attnw, attn_b, s.h0b, s.qf);
    attn_kernel<<<kB, 256, 0, stream>>>(x, t, context, s.qf, emb, s.inpb);
    gru_gates_kernel<TB><<<kGATES / 16 / 8, 256, 0, stream>>>(
        s.inpb, kIN0, wih0, bih0, s.h0b, whh0, bhh0, s.gi0, s.gh0);
    gru_combine_kernel<<<(kB * kHID) / 256, 256, 0, stream>>>(
        s.gi0, s.gh0, s.h0f, s.h0b, (__bf16*)nullptr, x, t);
    gru_gates_kernel<TB><<<kGATES / 16 / 8, 256, 0, stream>>>(
        s.h0b, kHID, wih1, bih1, s.h1b, whh1, bhh1, s.gi1, s.gh1);
    gru_combine_kernel<<<(kB * kHID) / 256, 256, 0, stream>>>(
        s.gi1, s.gh1, s.h1f, s.h1b, s.outb, x, t);
    logits_kernel<TB><<<kVOCAB / 16 / 8, 256, 0, stream>>>(s.outb, wout, bout, dout, t);
  }
}

extern "C" void kernel_launch(void* const* d_in, const int* in_sizes, int n_in,
                              void* d_out, int out_size, void* d_ws, size_t ws_size,
                              hipStream_t stream) {
  const int*   x       = (const int*)d_in[0];
  const float* context = (const float*)d_in[1];
  const float* emb     = (const float*)d_in[2];
  const float* attn_w  = (const float*)d_in[3];
  const float* attn_b  = (const float*)d_in[4];
  const float* w_ih0   = (const float*)d_in[5];
  const float* w_hh0   = (const float*)d_in[6];
  const float* b_ih0   = (const float*)d_in[7];
  const float* b_hh0   = (const float*)d_in[8];
  const float* w_ih1   = (const float*)d_in[9];
  const float* w_hh1   = (const float*)d_in[10];
  const float* b_ih1   = (const float*)d_in[11];
  const float* b_hh1   = (const float*)d_in[12];
  const float* w_out   = (const float*)d_in[13];
  const float* b_out   = (const float*)d_in[14];
  float* dout = (float*)d_out;

  char* ws = (char*)d_ws;
  size_t off = 0;
  auto take = [&](size_t bytes) -> char* {
    char* p = ws + off;
    off = (off + bytes + 255) & ~(size_t)255;
    return p;
  };

  Scratch s;
  s.h0f  = (float*)take((size_t)kB * kHID * 4);
  s.h1f  = (float*)take((size_t)kB * kHID * 4);
  s.h0b  = (__bf16*)take((size_t)kB * kHID * 2);
  s.h1b  = (__bf16*)take((size_t)kB * kHID * 2);
  s.qf   = (float*)take((size_t)kB * kCTX * 4);
  s.inpb = (__bf16*)take((size_t)kB * kIN0 * 2);
  s.gi0  = (float*)take((size_t)kB * kGATES * 4);
  s.gh0  = (float*)take((size_t)kB * kGATES * 4);
  s.gi1  = (float*)take((size_t)kB * kGATES * 4);
  s.gh1  = (float*)take((size_t)kB * kGATES * 4);
  s.outb = (__bf16*)take((size_t)kB * kHID * 2);
  size_t act_end = off;

  // bf16 weight shadows (all fit in the 192MB L2; halves per-step L2 traffic)
  __bf16* attnwb = (__bf16*)take((size_t)kCTX * (kEMB + kHID) * 2);
  __bf16* wih0b  = (__bf16*)take((size_t)kGATES * kIN0 * 2);
  __bf16* whh0b  = (__bf16*)take((size_t)kGATES * kHID * 2);
  __bf16* wih1b  = (__bf16*)take((size_t)kGATES * kHID * 2);
  __bf16* whh1b  = (__bf16*)take((size_t)kGATES * kHID * 2);
  __bf16* woutb  = (__bf16*)take((size_t)kVOCAB * kHID * 2);
  size_t full_end = off;

  if (ws_size >= full_end) {
    auto cvt = [&](const float* srcp, __bf16* dstp, size_t n) {
      int blocks = (int)((n + 255) / 256);
      if (blocks > 4096) blocks = 4096;
      cvt_bf16_kernel<<<blocks, 256, 0, stream>>>(srcp, dstp, n);
    };
    cvt(attn_w, attnwb, (size_t)kCTX * (kEMB + kHID));
    cvt(w_ih0, wih0b, (size_t)kGATES * kIN0);
    cvt(w_hh0, whh0b, (size_t)kGATES * kHID);
    cvt(w_ih1, wih1b, (size_t)kGATES * kHID);
    cvt(w_hh1, whh1b, (size_t)kGATES * kHID);
    cvt(w_out, woutb, (size_t)kVOCAB * kHID);
    run_model<__bf16>(x, context, emb, attnwb, attn_b,
                      wih0b, b_ih0, whh0b, b_hh0,
                      wih1b, b_ih1, whh1b, b_hh1,
                      woutb, b_out, s, dout, stream);
  } else if (ws_size >= act_end) {
    // fallback: convert f32 weights to bf16 fragments on the fly
    run_model<float>(x, context, emb, attn_w, attn_b,
                     w_ih0, b_ih0, w_hh0, b_hh0,
                     w_ih1, b_ih1, w_hh1, b_hh1,
                     w_out, b_out, s, dout, stream);
  }
  // else: workspace too small to run at all
}
